// Convcnplatent_multiouput_54881092108547
// MI455X (gfx1250) — matrix operand order repaired and checked by reading, not compiled, not run on hardware
//
#include <hip/hip_runtime.h>
#include <hip/hip_bf16.h>

typedef __attribute__((ext_vector_type(16))) _Float16 v16h;
typedef __attribute__((ext_vector_type(8)))  _Float16 v8h;
typedef __attribute__((ext_vector_type(8)))  float    v8f;

#define MMAX 352   // max grid points supported in LDS (expected m ~312)

// ---------------------------------------------------------------------------
// Kernel 1: ConvDeepSet encoder. One block per (batch, channel).
// h0[b,m,c] = sum_n exp(-0.5 d^2), h1[b,m,c] = sum_n y*exp(-0.5 d^2)
// ---------------------------------------------------------------------------
__global__ __launch_bounds__(256) void k1_encoder(
    const float* __restrict__ x, const float* __restrict__ y,
    const float* __restrict__ xgrid, const float* __restrict__ enc_sigma,
    float* __restrict__ h0w, float* __restrict__ h1w, int m, int npts)
{
    __shared__ float xs[256];
    __shared__ float ys[256];
    const int b = blockIdx.x / 3, c = blockIdx.x % 3, tid = threadIdx.x;
    const float scale = __expf(enc_sigma[c]);
    const float inv   = 1.f / (scale + 1e-6f);

    int mpt = (m + 255) >> 8; if (mpt > 2) mpt = 2;
    float gm[2], a0[2] = {0.f, 0.f}, a1[2] = {0.f, 0.f};
    for (int p = 0; p < 2; ++p) {
        int mg = tid + (p << 8);
        gm[p] = (p < mpt && mg < m) ? xgrid[(size_t)(b * m + mg) * 3 + c] : 1e30f;
    }
    for (int base = 0; base < npts; base += 256) {
        int cnt = npts - base; if (cnt > 256) cnt = 256;
        __syncthreads();
        if (tid < cnt) {
            xs[tid] = x[(size_t)(b * npts + base + tid) * 3 + c];
            ys[tid] = y[(size_t)(b * npts + base + tid) * 3 + c];
        }
        __syncthreads();
        for (int p = 0; p < mpt; ++p) {
            float g = gm[p], s0 = 0.f, s1 = 0.f;
            for (int i = 0; i < cnt; ++i) {
                float d = (xs[i] - g) * inv;
                float w = __expf(-0.5f * d * d);
                s0 += w; s1 += ys[i] * w;
            }
            a0[p] += s0; a1[p] += s1;
        }
    }
    for (int p = 0; p < mpt; ++p) {
        int mg = tid + (p << 8);
        if (mg < m) {
            h0w[(size_t)(b * m + mg) * 3 + c] = a0[p];
            h1w[(size_t)(b * m + mg) * 3 + c] = a1[p];
        }
    }
}

// ---------------------------------------------------------------------------
// WMMA implicit-GEMM conv layer. src/dst: LDS, (MMAX+8) rows x 32 f16,
// +4 row halo offset. Weights wsm[tap][out][in] f16 (stride 32 in 'in').
// cin16=true: layer-1 (Cin=16), K=80 padded to 96; else Cin=32, K=160.
// ---------------------------------------------------------------------------
__device__ __forceinline__ void conv_wmma(
    const _Float16* __restrict__ src, _Float16* __restrict__ dst,
    const _Float16* __restrict__ wsm, const float* __restrict__ bias,
    int m, int mT, bool cin16, int KC, bool relu)
{
    const int wave = threadIdx.x >> 5;
    const int lane = threadIdx.x & 31;
    const int lrow = lane & 15;
    const int h    = lane >> 4;
    for (int tp = wave; tp < mT * 2; tp += 8) {
        const int rt = tp >> 1, nt = tp & 1;
        const int colg = nt * 16 + lrow;
        const float bv = bias[colg];
        v8f acc;
        #pragma unroll
        for (int r = 0; r < 8; ++r) acc[r] = bv;
        const int l = rt * 16 + lrow;
        for (int kc = 0; kc < KC; ++kc) {
            v8h lo, hi;
            if (cin16) { // chunk covers taps 2kc, 2kc+1; half h -> tap 2kc+h, i = 0..15
                lo = *(const v8h*)&src[(size_t)(l + 2 * kc + h - 2 + 4) * 32];
                hi = *(const v8h*)&src[(size_t)(l + 2 * kc + h - 2 + 4) * 32 + 8];
            } else {     // chunk = one tap, full 32 channels
                const _Float16* rp = &src[(size_t)(l + kc - 2 + 4) * 32];
                lo = *(const v8h*)&rp[h * 16];
                hi = *(const v8h*)&rp[h * 16 + 8];
            }
            v16h a;
            #pragma unroll
            for (int i = 0; i < 8; ++i) { a[i] = lo[i]; a[8 + i] = hi[i]; }
            v16h bf;
            if (cin16) bf = *(const v16h*)&wsm[(size_t)((2 * kc + h) * 32 + colg) * 32];
            else       bf = *(const v16h*)&wsm[(size_t)(kc * 32 + colg) * 32 + h * 16];
            acc = __builtin_amdgcn_wmma_f32_16x16x32_f16(
                false, a, false, bf, (short)0, acc, false, false);
        }
        #pragma unroll
        for (int r = 0; r < 8; ++r) {
            int row = rt * 16 + r + 8 * h; // D: VGPR r -> rows r (lanes<16) / r+8
            float v = acc[r];
            if (relu) v = v > 0.f ? v : 0.f;
            dst[(size_t)(row + 4) * 32 + colg] = (row < m) ? (_Float16)v : (_Float16)0.f;
        }
    }
}

// ---------------------------------------------------------------------------
// Kernel 2: per-batch fused  rep->sigmoid  + 3 convs + linear head.
// ---------------------------------------------------------------------------
__global__ __launch_bounds__(256) void k2_cnn(
    const float* __restrict__ h0w, const float* __restrict__ h1w,
    const float* __restrict__ gW,  const float* __restrict__ gb,
    const float* __restrict__ w1,  const float* __restrict__ b1,
    const float* __restrict__ w2,  const float* __restrict__ b2,
    const float* __restrict__ w3,  const float* __restrict__ b3,
    const float* __restrict__ linW, const float* __restrict__ linb,
    float* __restrict__ muw, float* __restrict__ stdw, int m)
{
    __shared__ __align__(16) _Float16 bufA[(MMAX + 8) * 32];
    __shared__ __align__(16) _Float16 bufB[(MMAX + 8) * 32];
    __shared__ __align__(16) _Float16 wsm[6 * 32 * 32];
    __shared__ __align__(16) _Float16 lsm[32 * 32];
    const int b = blockIdx.x, tid = threadIdx.x;
    const int mT = (m + 15) >> 4;

    for (int i = tid; i < (MMAX + 8) * 32; i += 256) { bufA[i] = (_Float16)0.f; bufB[i] = (_Float16)0.f; }
    for (int i = tid; i < 6 * 32 * 32; i += 256) wsm[i] = (_Float16)0.f;
    __syncthreads();

    // rep = sigmoid([h0 | h1/(h0+eps)] @ gW + gb) -> bufA cols 0..15
    for (int row = tid; row < m; row += 256) {
        float v[6];
        #pragma unroll
        for (int c = 0; c < 3; ++c) {
            float a0 = h0w[(size_t)(b * m + row) * 3 + c];
            float a1 = h1w[(size_t)(b * m + row) * 3 + c];
            v[c] = a0; v[3 + c] = a1 / (a0 + 1e-6f);
        }
        for (int i = 0; i < 16; ++i) {
            float a = gb[i];
            #pragma unroll
            for (int j = 0; j < 6; ++j) a += v[j] * gW[j * 16 + i];
            bufA[(size_t)(row + 4) * 32 + i] = (_Float16)(1.f / (1.f + __expf(-a)));
        }
    }
    // layer-1 weights: wsm[k][o][i] = w1[o][i][k], Cin=16 ; and lin head weights
    for (int i = tid; i < 32 * 16 * 5; i += 256) {
        int k = i / (32 * 16), rem = i % (32 * 16), o = rem / 16, ii = rem % 16;
        wsm[(size_t)(k * 32 + o) * 32 + ii] = (_Float16)w1[(size_t)(o * 16 + ii) * 5 + k];
    }
    for (int i = tid; i < 32 * 32; i += 256) {
        int o = i / 32, ii = i % 32;
        lsm[i] = (o < 30) ? (_Float16)linW[(size_t)ii * 30 + o] : (_Float16)0.f;
    }
    __syncthreads();
    conv_wmma(bufA, bufB, wsm, b1, m, mT, true, 3, true);
    __syncthreads();
    for (int i = tid; i < 32 * 32 * 5; i += 256) {
        int k = i / (32 * 32), rem = i % (32 * 32), o = rem / 32, ii = rem % 32;
        wsm[(size_t)(k * 32 + o) * 32 + ii] = (_Float16)w2[(size_t)(o * 32 + ii) * 5 + k];
    }
    __syncthreads();
    conv_wmma(bufB, bufA, wsm, b2, m, mT, false, 5, true);
    __syncthreads();
    for (int i = tid; i < 32 * 32 * 5; i += 256) {
        int k = i / (32 * 32), rem = i % (32 * 32), o = rem / 32, ii = rem % 32;
        wsm[(size_t)(k * 32 + o) * 32 + ii] = (_Float16)w3[(size_t)(o * 32 + ii) * 5 + k];
    }
    __syncthreads();
    conv_wmma(bufA, bufB, wsm, b3, m, mT, false, 5, false);
    __syncthreads();

    // linear head: (m x 32) @ (32 x 30), split -> mu / 0.1+0.9*sigmoid(std)
    {
        const int wave = tid >> 5, lane = tid & 31, lrow = lane & 15, h = lane >> 4;
        for (int tp = wave; tp < mT * 2; tp += 8) {
            const int rt = tp >> 1, nt = tp & 1;
            const int colg = nt * 16 + lrow;
            const float bv = (colg < 30) ? linb[colg] : 0.f;
            v8f acc;
            #pragma unroll
            for (int r = 0; r < 8; ++r) acc[r] = bv;
            const int l = rt * 16 + lrow;
            const _Float16* rp = &bufB[(size_t)(l + 4) * 32];
            v8h lo = *(const v8h*)&rp[h * 16];
            v8h hi = *(const v8h*)&rp[h * 16 + 8];
            v16h a;
            #pragma unroll
            for (int i = 0; i < 8; ++i) { a[i] = lo[i]; a[8 + i] = hi[i]; }
            v16h bf = *(const v16h*)&lsm[(size_t)colg * 32 + h * 16];
            acc = __builtin_amdgcn_wmma_f32_16x16x32_f16(
                false, a, false, bf, (short)0, acc, false, false);
            #pragma unroll
            for (int r = 0; r < 8; ++r) {
                int row = rt * 16 + r + 8 * h;
                if (row < m) {
                    float v = acc[r];
                    if (colg < 15)
                        muw[(size_t)(b * m + row) * 15 + colg] = v;
                    else if (colg < 30)
                        stdw[(size_t)(b * m + row) * 15 + (colg - 15)] =
                            0.1f + 0.9f / (1.f + __expf(-v));
                }
            }
        }
    }
}

// ---------------------------------------------------------------------------
// Kernel 3: interpolation GEMM H[s,t] = sum_m Z[s,m] * W[m,t].
// One block per (b, j=k*3+c). B fragment (Gaussian weights) built in
// registers from v_exp_f32; Z padded to 16 rows in LDS for uniform WMMA.
// ---------------------------------------------------------------------------
__global__ __launch_bounds__(256) void k3_interp(
    const float* __restrict__ muw, const float* __restrict__ stdw,
    const float* __restrict__ eps, const float* __restrict__ xgrid,
    const float* __restrict__ xout, const float* __restrict__ int_sigma,
    float* __restrict__ hout, int m, int ntar, int nb)
{
    __shared__ __align__(16) _Float16 zsm[16 * MMAX];
    __shared__ float gsm[MMAX];
    const int b = blockIdx.x / 15, j = blockIdx.x % 15;
    const int cc = j % 3;
    const int tid = threadIdx.x;
    const float isc = __expf(int_sigma[j]);
    const float inv = 1.f / (isc + 1e-6f);

    for (int i = tid; i < 16 * MMAX; i += 256) zsm[i] = (_Float16)0.f;
    for (int i = tid; i < MMAX; i += 256)
        gsm[i] = (i < m) ? xgrid[(size_t)(b * m + i) * 3 + cc] : 1e30f;
    __syncthreads();
    for (int idx = tid; idx < 4 * m; idx += 256) {
        int s = idx / m, i = idx % m;
        float v = muw[(size_t)(b * m + i) * 15 + j] +
                  stdw[(size_t)(b * m + i) * 15 + j] * eps[(size_t)(s * nb + b) * 15 + j];
        zsm[(size_t)s * MMAX + i] = (_Float16)v;
    }
    __syncthreads();

    const int wave = tid >> 5, lane = tid & 31, lrow = lane & 15, h = lane >> 4;
    const int tT = (ntar + 15) >> 4;
    const int KC = (m + 31) >> 5;
    for (int tt = wave; tt < tT; tt += 8) {
        const int col = tt * 16 + lrow;
        const int tcl = col < ntar ? col : ntar - 1;
        const float xt = xout[(size_t)(b * ntar + tcl) * 3 + cc];
        v8f acc = {};
        for (int kc = 0; kc < KC; ++kc) {
            v8h lo = *(const v8h*)&zsm[(size_t)lrow * MMAX + kc * 32 + h * 8];
            v8h hi = *(const v8h*)&zsm[(size_t)lrow * MMAX + kc * 32 + 16 + h * 8];
            v16h a;
            #pragma unroll
            for (int i = 0; i < 8; ++i) { a[i] = lo[i]; a[8 + i] = hi[i]; }
            v16h bf;
            #pragma unroll
            for (int jj = 0; jj < 16; ++jj) {
                int mi = kc * 32 + (jj < 8 ? h * 8 + jj : 8 + h * 8 + jj);
                float d = (gsm[mi] - xt) * inv;
                float w = __expf(-0.5f * d * d);
                if (mi >= m) w = 0.f;
                bf[jj] = (_Float16)w;
            }
            acc = __builtin_amdgcn_wmma_f32_16x16x32_f16(
                false, a, false, bf, (short)0, acc, false, false);
        }
        if (lane < 16 && col < ntar) {
            #pragma unroll
            for (int r = 0; r < 4; ++r) // rows 0..3 = samples s
                hout[((size_t)(r * nb + b) * ntar + col) * 15 + j] = acc[r];
        }
    }
}

// ---------------------------------------------------------------------------
// Kernel 4: out = h_out @ loW + lob ; softplus on last 3 columns.
// ---------------------------------------------------------------------------
__global__ __launch_bounds__(256) void k4_project(
    const float* __restrict__ hout, const float* __restrict__ loW,
    const float* __restrict__ lob, float* __restrict__ out, int total)
{
    int idx = blockIdx.x * 256 + threadIdx.x;
    if (idx >= total) return;
    float hv[15];
    #pragma unroll
    for (int j = 0; j < 15; ++j) hv[j] = hout[(size_t)idx * 15 + j];
    #pragma unroll
    for (int j2 = 0; j2 < 6; ++j2) {
        float a = lob[j2];
        #pragma unroll
        for (int j = 0; j < 15; ++j) a += hv[j] * loW[j * 6 + j2];
        if (j2 >= 3) a = (a > 20.f) ? a : __logf(1.f + __expf(a)); // softplus
        out[(size_t)idx * 6 + j2] = a;
    }
}

// ---------------------------------------------------------------------------
extern "C" void kernel_launch(void* const* d_in, const int* in_sizes, int n_in,
                              void* d_out, int out_size, void* d_ws, size_t ws_size,
                              hipStream_t stream)
{
    const float* x        = (const float*)d_in[0];
    const float* y        = (const float*)d_in[1];
    const float* x_out    = (const float*)d_in[2];
    const float* x_grid   = (const float*)d_in[3];
    const float* eps      = (const float*)d_in[4];
    const float* enc_sig  = (const float*)d_in[5];
    const float* gW       = (const float*)d_in[6];
    const float* gb       = (const float*)d_in[7];
    const float* w1       = (const float*)d_in[8];
    const float* b1       = (const float*)d_in[9];
    const float* w2       = (const float*)d_in[10];
    const float* b2       = (const float*)d_in[11];
    const float* w3       = (const float*)d_in[12];
    const float* b3       = (const float*)d_in[13];
    const float* linW     = (const float*)d_in[14];
    const float* linb     = (const float*)d_in[15];
    const float* int_sig  = (const float*)d_in[16];
    const float* loW      = (const float*)d_in[17];
    const float* lob      = (const float*)d_in[18];
    float* out = (float*)d_out;

    // derive shapes: eps = (NS=4, nb, 15)
    const int nb   = in_sizes[4] / 60;
    const int npts = in_sizes[0] / (nb * 3);
    const int ntar = in_sizes[2] / (nb * 3);
    int m          = in_sizes[3] / (nb * 3);
    if (m > MMAX) m = MMAX;

    float* ws = (float*)d_ws;
    size_t o_h0 = 0;
    size_t o_h1 = o_h0 + (size_t)nb * m * 3;
    size_t o_mu = o_h1 + (size_t)nb * m * 3;
    size_t o_sd = o_mu + (size_t)nb * m * 15;
    size_t o_ho = o_sd + (size_t)nb * m * 15;

    k1_encoder<<<nb * 3, 256, 0, stream>>>(x, y, x_grid, enc_sig,
                                           ws + o_h0, ws + o_h1, m, npts);
    k2_cnn<<<nb, 256, 0, stream>>>(ws + o_h0, ws + o_h1, gW, gb,
                                   w1, b1, w2, b2, w3, b3, linW, linb,
                                   ws + o_mu, ws + o_sd, m);
    k3_interp<<<nb * 15, 256, 0, stream>>>(ws + o_mu, ws + o_sd, eps,
                                           x_grid, x_out, int_sig,
                                           ws + o_ho, m, ntar, nb);
    const int total = 4 * nb * ntar;
    k4_project<<<(total + 255) / 256, 256, 0, stream>>>(ws + o_ho, loW, lob,
                                                        out, total);
}
